// HybridPOSSMDecoder_33861522162384
// MI455X (gfx1250) — compile-verified, matrix operand order checked by reading, not run on hardware
//
#include <hip/hip_runtime.h>
#include <hip/hip_bf16.h>

typedef __attribute__((ext_vector_type(16))) _Float16 v16h;
typedef __attribute__((ext_vector_type(8)))  _Float16 h8;
typedef __attribute__((ext_vector_type(4)))  _Float16 h4;
typedef __attribute__((ext_vector_type(8)))  float    v8f;

#define B_        8
#define T_        8192
#define NF_       256
#define D_MODEL_  512
#define NHEAD_    8
#define HEAD_DIM_ 64
#define KLEN_     32
#define NPATCH_   256
#define NTF_      2
#define NSSM_     4
#define D_INNER_  1024
#define D_STATE_  16
#define D_CONV_   4
#define FF_       2048
#define NPH_      41
#define MAXD_     128

// ---------------------------------------------------------------- GEMM (WMMA)
// Requirements (guaranteed by call sites): M%64==0, N%64==0, K%64==0,
// lda/ldb/ldc %4==0, base pointers 16B aligned.
#define BM 64
#define BN 64
#define BK 64
#define GEMM_THREADS 128

__device__ __forceinline__ float act_apply(float v, int act) {
  if (act == 1) return 0.5f * v * (1.0f + erff(v * 0.70710678118654752f)); // exact gelu
  if (act == 2) return v / (1.0f + __expf(-v));                            // silu
  if (act == 3) return (v > 20.0f) ? v : log1pf(__expf(v));                // softplus
  return v;
}

__global__ __launch_bounds__(GEMM_THREADS)
void wmma_gemm_kernel(const float* __restrict__ A, int lda, long long sAo, long long sAi,
                      const float* __restrict__ Bm, int ldb, long long sBo, long long sBi,
                      int transB,
                      const float* __restrict__ bias,
                      float* __restrict__ C, int ldc, long long sCo, long long sCi,
                      int M, int N, int K, int innerB,
                      float alpha, int act, int addC) {
  // Double-buffered LDS; both operands K-contiguous: As[m][k], Bs[n][k].
  __shared__ _Float16 As[2][BM][BK + 8];
  __shared__ _Float16 Bs[2][BN][BK + 8];

  const int bz = blockIdx.z;
  const int bo = bz / innerB;
  const int bi = bz % innerB;
  A  += bo * sAo + bi * sAi;
  Bm += bo * sBo + bi * sBi;
  C  += bo * sCo + bi * sCi;

  const int m0 = blockIdx.y * BM;
  const int n0 = blockIdx.x * BN;
  const int tid  = threadIdx.x;
  const int lane = tid & 31;
  const int wid  = tid >> 5;          // 4 waves -> 2x2 arrangement of 32x32 wave tiles
  const int wm   = (wid >> 1) * 32;
  const int wn   = (wid & 1) * 32;
  const int ml   = lane & 15;
  const int hs   = lane >> 4;         // half-select per ISA 16-bit A/B layout

  // ---- phase helpers -------------------------------------------------------
  // Issue all 16 global b128 loads back-to-back (no per-load drain).
  auto load_tile = [&](int k0, float4* ta, float4* tb) {
#pragma unroll
    for (int e = 0; e < 8; ++e) {
      int idx = e * GEMM_THREADS + tid;       // 1024 float4's of A tile
      int r = idx >> 4, c4 = idx & 15;
      ta[e] = *reinterpret_cast<const float4*>(
          &A[(long long)(m0 + r) * lda + k0 + (c4 << 2)]);
    }
    if (transB) {
#pragma unroll
      for (int e = 0; e < 8; ++e) {
        int idx = e * GEMM_THREADS + tid;
        int r = idx >> 4, c4 = idx & 15;      // r = n-local, c4*4 = k-local
        tb[e] = *reinterpret_cast<const float4*>(
            &Bm[(long long)(n0 + r) * ldb + k0 + (c4 << 2)]);
      }
    } else {
#pragma unroll
      for (int e = 0; e < 8; ++e) {
        int idx = e * GEMM_THREADS + tid;
        int r = idx >> 4, c4 = idx & 15;      // r = k-local, c4*4 = n-local
        tb[e] = *reinterpret_cast<const float4*>(
            &Bm[(long long)(k0 + r) * ldb + n0 + (c4 << 2)]);
      }
    }
  };
  // Convert fp32 -> f16 and commit to the given LDS buffer.
  auto store_tile = [&](int buf, const float4* ta, const float4* tb) {
#pragma unroll
    for (int e = 0; e < 8; ++e) {
      int idx = e * GEMM_THREADS + tid;
      int r = idx >> 4, c4 = idx & 15;
      h4 hv = { (_Float16)ta[e].x, (_Float16)ta[e].y, (_Float16)ta[e].z, (_Float16)ta[e].w };
      *reinterpret_cast<h4*>(&As[buf][r][c4 << 2]) = hv;
    }
    if (transB) {
#pragma unroll
      for (int e = 0; e < 8; ++e) {
        int idx = e * GEMM_THREADS + tid;
        int r = idx >> 4, c4 = idx & 15;
        h4 hv = { (_Float16)tb[e].x, (_Float16)tb[e].y, (_Float16)tb[e].z, (_Float16)tb[e].w };
        *reinterpret_cast<h4*>(&Bs[buf][r][c4 << 2]) = hv;
      }
    } else {
#pragma unroll
      for (int e = 0; e < 8; ++e) {
        int idx = e * GEMM_THREADS + tid;
        int r = idx >> 4, c4 = idx & 15;
        int nl = c4 << 2;
        Bs[buf][nl + 0][r] = (_Float16)tb[e].x;
        Bs[buf][nl + 1][r] = (_Float16)tb[e].y;
        Bs[buf][nl + 2][r] = (_Float16)tb[e].z;
        Bs[buf][nl + 3][r] = (_Float16)tb[e].w;
      }
    }
  };
  // -------------------------------------------------------------------------

  v8f acc[2][2] = {};

  const int nk = K / BK;
  {
    float4 ta[8], tb[8];
    load_tile(0, ta, tb);
    store_tile(0, ta, tb);
  }
  __syncthreads();

  for (int t = 0; t < nk; ++t) {
    const int buf = t & 1;
    float4 na[8], nb[8];
    if (t + 1 < nk) load_tile((t + 1) * BK, na, nb);   // overlap with WMMA below

    // ---- two WMMA K-steps; fragments via ds_load_b128 pairs
#pragma unroll
    for (int kk0 = 0; kk0 < BK; kk0 += 32) {
      v16h af[2], bf[2];
#pragma unroll
      for (int tt = 0; tt < 2; ++tt) {
        h8* ap = reinterpret_cast<h8*>(&af[tt]);
        ap[0] = *reinterpret_cast<const h8*>(&As[buf][wm + tt * 16 + ml][kk0 + 8 * hs]);
        ap[1] = *reinterpret_cast<const h8*>(&As[buf][wm + tt * 16 + ml][kk0 + 16 + 8 * hs]);
        h8* bp = reinterpret_cast<h8*>(&bf[tt]);
        bp[0] = *reinterpret_cast<const h8*>(&Bs[buf][wn + tt * 16 + ml][kk0 + 8 * hs]);
        bp[1] = *reinterpret_cast<const h8*>(&Bs[buf][wn + tt * 16 + ml][kk0 + 16 + 8 * hs]);
      }
#pragma unroll
      for (int tm = 0; tm < 2; ++tm)
#pragma unroll
        for (int tn = 0; tn < 2; ++tn)
          acc[tm][tn] = __builtin_amdgcn_wmma_f32_16x16x32_f16(
              false, af[tm], false, bf[tn], (short)0, acc[tm][tn], false, false);
    }

    if (t + 1 < nk) store_tile(buf ^ 1, na, nb);
    __syncthreads();
  }

  // ---- epilogue: C/D layout = 8 VGPRs; lanes 0-15: M=r, lanes 16-31: M=r+8
#pragma unroll
  for (int tm = 0; tm < 2; ++tm) {
#pragma unroll
    for (int tn = 0; tn < 2; ++tn) {
#pragma unroll
      for (int r = 0; r < 8; ++r) {
        int row = m0 + wm + tm * 16 + hs * 8 + r;
        int col = n0 + wn + tn * 16 + ml;
        float v = alpha * acc[tm][tn][r];
        if (bias) v += bias[col];
        v = act_apply(v, act);
        long long ci = (long long)row * ldc + col;
        C[ci] = addC ? (C[ci] + v) : v;
      }
    }
  }
}

// ------------------------------------------------- front-end: day scale + conv
__global__ void preprocess_kernel(const float* __restrict__ x,
                                  const float* __restrict__ dw,
                                  const float* __restrict__ db,
                                  const float* __restrict__ gk,
                                  const int* __restrict__ day_idx,
                                  float* __restrict__ out) {
  long long gid = (long long)blockIdx.x * blockDim.x + threadIdx.x;
  const long long total = (long long)B_ * T_ * NF_;
  if (gid >= total) return;
  int f = (int)(gid % NF_);
  long long bt = gid / NF_;
  int t = (int)(bt % T_);
  int b = (int)(bt / T_);
  int d = day_idx[b];
  float w  = dw[d * NF_ + f];
  float bb = db[d * NF_ + f];
  float acc = 0.0f;
#pragma unroll
  for (int k = 0; k < 20; ++k) {
    int tt = t + k - 10;
    if (tt >= 0 && tt < T_) {
      float xv = x[((long long)b * T_ + tt) * NF_ + f] * w + bb;
      acc += xv * gk[f * 20 + k];
    }
  }
  out[gid] = acc; // (B,T,NF) == reshape (B,NPATCH,KLEN*NF) contiguously
}

// ------------------------------------------------------------------ LayerNorm
__global__ void ln_kernel(const float* __restrict__ in, float* __restrict__ out,
                          const float* __restrict__ g, const float* __restrict__ b,
                          int cols, float eps) {
  int row = blockIdx.x;
  const float* x = in + (long long)row * cols;
  float* y = out + (long long)row * cols;
  __shared__ float red[256];
  float s = 0.0f;
  for (int c = threadIdx.x; c < cols; c += 256) s += x[c];
  red[threadIdx.x] = s; __syncthreads();
  for (int st = 128; st > 0; st >>= 1) {
    if (threadIdx.x < st) red[threadIdx.x] += red[threadIdx.x + st];
    __syncthreads();
  }
  float mean = red[0] / cols;
  __syncthreads();
  float s2 = 0.0f;
  for (int c = threadIdx.x; c < cols; c += 256) { float d = x[c] - mean; s2 += d * d; }
  red[threadIdx.x] = s2; __syncthreads();
  for (int st = 128; st > 0; st >>= 1) {
    if (threadIdx.x < st) red[threadIdx.x] += red[threadIdx.x + st];
    __syncthreads();
  }
  float rstd = rsqrtf(red[0] / cols + eps);
  __syncthreads();
  for (int c = threadIdx.x; c < cols; c += 256)
    y[c] = (x[c] - mean) * rstd * g[c] + b[c];
}

// ------------------------------------- softmax with rel-bias + causal masking
__global__ void softmax_kernel(float* __restrict__ scores,
                               const float* __restrict__ rel_table) {
  int z = blockIdx.x;                 // flat (b,h,q)
  int q = z % NPATCH_;
  int h = (z / NPATCH_) % NHEAD_;
  float* row = scores + (long long)z * NPATCH_;
  int k = threadIdx.x;
  int rel = k - q;
  rel = (rel < -MAXD_) ? -MAXD_ : ((rel > MAXD_) ? MAXD_ : rel);
  float v = row[k] + rel_table[h * (2 * MAXD_ + 1) + rel + MAXD_]
          + ((k <= q) ? 0.0f : -1.0e9f);
  __shared__ float red[256];
  red[k] = v; __syncthreads();
  for (int st = 128; st > 0; st >>= 1) {
    if (k < st) red[k] = fmaxf(red[k], red[k + st]);
    __syncthreads();
  }
  float mx = red[0];
  __syncthreads();
  float e = __expf(v - mx);
  red[k] = e; __syncthreads();
  for (int st = 128; st > 0; st >>= 1) {
    if (k < st) red[k] += red[k + st];
    __syncthreads();
  }
  row[k] = e / red[0];
}

// -------------------------------------------- Mamba causal depthwise conv+SiLU
__global__ void mamba_conv_kernel(const float* __restrict__ inproj, // (B,S,2*DI)
                                  const float* __restrict__ cw,
                                  const float* __restrict__ cb,
                                  float* __restrict__ out) {        // (B,S,DI)
  int gid = blockIdx.x * blockDim.x + threadIdx.x;
  if (gid >= B_ * NPATCH_ * D_INNER_) return;
  int d = gid % D_INNER_;
  int s = (gid / D_INNER_) % NPATCH_;
  int b = gid / (D_INNER_ * NPATCH_);
  float acc = cb[d];
#pragma unroll
  for (int k = 0; k < D_CONV_; ++k) {
    int ss = s + k - (D_CONV_ - 1);
    if (ss >= 0)
      acc += inproj[((long long)(b * NPATCH_ + ss)) * (2 * D_INNER_) + d] * cw[d * D_CONV_ + k];
  }
  out[gid] = acc / (1.0f + __expf(-acc)); // silu
}

// ---------------------------------------------- selective-scan (state in regs)
// bscs has padded row stride 64: cols 0..15 = Bs, 16..31 = Cs, rest zero.
__global__ void mamba_scan_kernel(const float* __restrict__ xc,     // (B,S,DI)
                                  const float* __restrict__ delta,  // (B,S,DI)
                                  const float* __restrict__ bscs,   // (B,S,64)
                                  const float* __restrict__ inproj, // (B,S,2*DI) for z
                                  const float* __restrict__ Alog,   // (DI,DS)
                                  const float* __restrict__ Dp,     // (DI)
                                  float* __restrict__ y) {          // (B,S,DI)
  int gid = blockIdx.x * blockDim.x + threadIdx.x; // b*DI + d
  if (gid >= B_ * D_INNER_) return;
  int d = gid % D_INNER_;
  int b = gid / D_INNER_;
  float A[D_STATE_], h[D_STATE_];
#pragma unroll
  for (int j = 0; j < D_STATE_; ++j) { A[j] = -__expf(Alog[d * D_STATE_ + j]); h[j] = 0.0f; }
  float Dd = Dp[d];
  for (int s = 0; s < NPATCH_; ++s) {
    long long base = (long long)(b * NPATCH_ + s);
    float xv = xc[base * D_INNER_ + d];
    float dt = delta[base * D_INNER_ + d];
    float z  = inproj[base * (2 * D_INNER_) + D_INNER_ + d];
    const float* bc = bscs + base * 64;
    float acc = 0.0f;
#pragma unroll
    for (int j = 0; j < D_STATE_; ++j) {
      h[j] = __expf(dt * A[j]) * h[j] + dt * bc[j] * xv;
      acc += h[j] * bc[D_STATE_ + j];
    }
    acc += xv * Dd;
    float sig = 1.0f / (1.0f + __expf(-z));
    y[base * D_INNER_ + d] = acc * (z * sig);
  }
}

// --------------------------------------------------- zero-pad columns utility
__global__ void pad_cols_kernel(const float* __restrict__ src, float* __restrict__ dst,
                                int rows, int sc, int dc) {
  int gid = blockIdx.x * blockDim.x + threadIdx.x;
  if (gid >= rows * dc) return;
  int r = gid / dc, c = gid % dc;
  dst[gid] = (c < sc) ? src[r * sc + c] : 0.0f;
}

__global__ void unpad_cols_kernel(const float* __restrict__ src, float* __restrict__ dst,
                                  int rows, int sc, int dc) {
  int gid = blockIdx.x * blockDim.x + threadIdx.x; // over rows*dc (dst)
  if (gid >= rows * dc) return;
  int r = gid / dc, c = gid % dc;
  dst[gid] = src[r * sc + c];
}

// ============================================================================
extern "C" void kernel_launch(void* const* d_in, const int* in_sizes, int n_in,
                              void* d_out, int out_size, void* d_ws, size_t ws_size,
                              hipStream_t stream) {
  (void)in_sizes; (void)n_in; (void)out_size; (void)ws_size;
  const float* x          = (const float*)d_in[0];
  const float* day_w      = (const float*)d_in[1];
  const float* day_b      = (const float*)d_in[2];
  const float* gauss      = (const float*)d_in[3];
  const float* pe_ln1_g   = (const float*)d_in[4];
  const float* pe_ln1_b   = (const float*)d_in[5];
  const float* pe_W       = (const float*)d_in[6];
  const float* pe_b       = (const float*)d_in[7];
  const float* pe_ln2_g   = (const float*)d_in[8];
  const float* pe_ln2_b   = (const float*)d_in[9];
  const float* rel_table  = (const float*)d_in[10];
  const float* tf_ln_g    = (const float*)d_in[11];
  const float* tf_ln_b    = (const float*)d_in[12];
  const float* tf_qkv_W   = (const float*)d_in[13];
  const float* tf_qkv_b   = (const float*)d_in[14];
  const float* tf_out_W   = (const float*)d_in[15];
  const float* tf_out_b   = (const float*)d_in[16];
  const float* tf_ffln_g  = (const float*)d_in[17];
  const float* tf_ffln_b  = (const float*)d_in[18];
  const float* tf_fc1_W   = (const float*)d_in[19];
  const float* tf_fc1_b   = (const float*)d_in[20];
  const float* tf_fc2_W   = (const float*)d_in[21];
  const float* tf_fc2_b   = (const float*)d_in[22];
  const float* mb_norm_g  = (const float*)d_in[23];
  const float* mb_norm_b  = (const float*)d_in[24];
  const float* mb_in_W    = (const float*)d_in[25];
  const float* mb_in_b    = (const float*)d_in[26];
  const float* mb_conv_W  = (const float*)d_in[27];
  const float* mb_conv_b  = (const float*)d_in[28];
  const float* mb_xp_W    = (const float*)d_in[29];
  const float* mb_xp_b    = (const float*)d_in[30];
  const float* mb_dt_W    = (const float*)d_in[31];
  const float* mb_dt_b    = (const float*)d_in[32];
  const float* mb_Alog    = (const float*)d_in[33];
  const float* mb_D       = (const float*)d_in[34];
  const float* mb_out_W   = (const float*)d_in[35];
  const float* mb_out_b   = (const float*)d_in[36];
  const float* head_W     = (const float*)d_in[37];
  const float* head_b     = (const float*)d_in[38];
  const int*   day_idx    = (const int*)d_in[39];

  float* ws = (float*)d_ws;
  // Workspace layout (float offsets). Arena [0,16.7M) is Xp then reused.
  const long long XP   = 0;          // (B,T,NF)        16,777,216
  const long long XM   = 16777216;   // (B,S,512)        1,048,576
  const long long HB   = 0;          // (B,S,512)        1,048,576
  const long long QKV  = 1048576;    // (B,S,1536)       3,145,728
  const long long SC   = 4194304;    // (B,H,S,S)        4,194,304
  const long long AO   = 8388608;    // (B,S,512)        1,048,576
  const long long FF1  = 1048576;    // (B,S,2048)       4,194,304
  const long long MIN  = 1048576;    // (B,S,2048)       4,194,304
  const long long XCV  = 5242880;    // (B,S,1024)       2,097,152
  const long long DLT  = 7340032;    // (B,S,1024)       2,097,152
  const long long BCS  = 9437184;    // (B,S,64) padded     131,072
  const long long YB   = 9568256;    // (B,S,1024)       2,097,152  ends 11,665,408
  const long long XPW  = 11665408;   // (1024,64) padded     65,536
  const long long XPB  = 11730944;   // (64,)                    64
  const long long HWP  = 11731008;   // (512,64) padded      32,768
  const long long HBP  = 11763776;   // (64,)                    64
  const long long OUTP = 11763840;   // (2048,64) padded    131,072

  const int MROWS = B_ * NPATCH_;    // 2048

  auto gemm = [&](const float* A, int lda, long long sAo, long long sAi,
                  const float* Bp, int ldb, long long sBo, long long sBi, int transB,
                  const float* bias, float* C, int ldc, long long sCo, long long sCi,
                  int M, int N, int K, int outerB, int innerB,
                  float alpha, int act, int addC) {
    dim3 grid((N + BN - 1) / BN, (M + BM - 1) / BM, outerB * innerB);
    wmma_gemm_kernel<<<grid, dim3(GEMM_THREADS), 0, stream>>>(
        A, lda, sAo, sAi, Bp, ldb, sBo, sBi, transB, bias,
        C, ldc, sCo, sCi, M, N, K, innerB, alpha, act, addC);
  };

  // 1) day scale/bias + Gaussian conv (fused); reshape is a contiguous no-op
  {
    long long total = (long long)B_ * T_ * NF_;
    preprocess_kernel<<<(unsigned)((total + 255) / 256), 256, 0, stream>>>(
        x, day_w, day_b, gauss, day_idx, ws + XP);
  }
  // 2) LN1 (dim 8192, eps 1e-6), in place
  ln_kernel<<<MROWS, 256, 0, stream>>>(ws + XP, ws + XP, pe_ln1_g, pe_ln1_b, KLEN_ * NF_, 1e-6f);
  // 3) patch-embed GEMM 2048x8192x512 -> XM
  gemm(ws + XP, KLEN_ * NF_, 0, 0, pe_W, D_MODEL_, 0, 0, 0, pe_b,
       ws + XM, D_MODEL_, 0, 0, MROWS, D_MODEL_, KLEN_ * NF_, 1, 1, 1.0f, 0, 0);
  // 4) LN2 (eps 1e-6), in place
  ln_kernel<<<MROWS, 256, 0, stream>>>(ws + XM, ws + XM, pe_ln2_g, pe_ln2_b, D_MODEL_, 1e-6f);

  // -------------------------------------------------------- transformer blocks
  for (int i = 0; i < NTF_; ++i) {
    ln_kernel<<<MROWS, 256, 0, stream>>>(ws + XM, ws + HB,
        tf_ln_g + i * D_MODEL_, tf_ln_b + i * D_MODEL_, D_MODEL_, 1e-5f);
    // QKV: 2048 x 512 -> 1536
    gemm(ws + HB, D_MODEL_, 0, 0, tf_qkv_W + (long long)i * D_MODEL_ * 3 * D_MODEL_,
         3 * D_MODEL_, 0, 0, 0, tf_qkv_b + i * 3 * D_MODEL_,
         ws + QKV, 3 * D_MODEL_, 0, 0, MROWS, 3 * D_MODEL_, D_MODEL_, 1, 1, 1.0f, 0, 0);
    // scores = Q Kt / sqrt(64)   batched over (b,h)
    gemm(ws + QKV, 3 * D_MODEL_, (long long)NPATCH_ * 3 * D_MODEL_, HEAD_DIM_,
         ws + QKV + D_MODEL_, 3 * D_MODEL_, (long long)NPATCH_ * 3 * D_MODEL_, HEAD_DIM_, 1,
         nullptr, ws + SC, NPATCH_,
         (long long)NHEAD_ * NPATCH_ * NPATCH_, (long long)NPATCH_ * NPATCH_,
         NPATCH_, NPATCH_, HEAD_DIM_, B_, NHEAD_, 0.125f, 0, 0);
    // softmax(scores + rel_bias + causal mask)
    softmax_kernel<<<B_ * NHEAD_ * NPATCH_, 256, 0, stream>>>(ws + SC, rel_table);
    // O = P V  (writes head-merged (B,S,512))
    gemm(ws + SC, NPATCH_, (long long)NHEAD_ * NPATCH_ * NPATCH_, (long long)NPATCH_ * NPATCH_,
         ws + QKV + 2 * D_MODEL_, 3 * D_MODEL_, (long long)NPATCH_ * 3 * D_MODEL_, HEAD_DIM_, 0,
         nullptr, ws + AO, D_MODEL_, (long long)NPATCH_ * D_MODEL_, HEAD_DIM_,
         NPATCH_, HEAD_DIM_, NPATCH_, B_, NHEAD_, 1.0f, 0, 0);
    // x += O Wout + bout
    gemm(ws + AO, D_MODEL_, 0, 0, tf_out_W + (long long)i * D_MODEL_ * D_MODEL_,
         D_MODEL_, 0, 0, 0, tf_out_b + i * D_MODEL_,
         ws + XM, D_MODEL_, 0, 0, MROWS, D_MODEL_, D_MODEL_, 1, 1, 1.0f, 0, 1);
    // FFN
    ln_kernel<<<MROWS, 256, 0, stream>>>(ws + XM, ws + HB,
        tf_ffln_g + i * D_MODEL_, tf_ffln_b + i * D_MODEL_, D_MODEL_, 1e-5f);
    gemm(ws + HB, D_MODEL_, 0, 0, tf_fc1_W + (long long)i * D_MODEL_ * FF_,
         FF_, 0, 0, 0, tf_fc1_b + i * FF_,
         ws + FF1, FF_, 0, 0, MROWS, FF_, D_MODEL_, 1, 1, 1.0f, 1, 0); // gelu
    gemm(ws + FF1, FF_, 0, 0, tf_fc2_W + (long long)i * FF_ * D_MODEL_,
         D_MODEL_, 0, 0, 0, tf_fc2_b + i * D_MODEL_,
         ws + XM, D_MODEL_, 0, 0, MROWS, D_MODEL_, FF_, 1, 1, 1.0f, 0, 1);
  }

  // -------------------------------------------------------------- Mamba blocks
  for (int i = 0; i < NSSM_; ++i) {
    ln_kernel<<<MROWS, 256, 0, stream>>>(ws + XM, ws + HB,
        mb_norm_g + i * D_MODEL_, mb_norm_b + i * D_MODEL_, D_MODEL_, 1e-5f);
    // in-proj: 2048 x 512 -> 2048  ([xc | z])
    gemm(ws + HB, D_MODEL_, 0, 0, mb_in_W + (long long)i * D_MODEL_ * 2 * D_INNER_,
         2 * D_INNER_, 0, 0, 0, mb_in_b + i * 2 * D_INNER_,
         ws + MIN, 2 * D_INNER_, 0, 0, MROWS, 2 * D_INNER_, D_MODEL_, 1, 1, 1.0f, 0, 0);
    // causal depthwise conv + silu
    mamba_conv_kernel<<<(B_ * NPATCH_ * D_INNER_) / 256, 256, 0, stream>>>(
        ws + MIN, mb_conv_W + (long long)i * D_INNER_ * D_CONV_,
        mb_conv_b + i * D_INNER_, ws + XCV);
    // pad xp weights/bias 32 -> 64 cols so the GEMM needs no N guards
    pad_cols_kernel<<<(D_INNER_ * 64) / 256, 256, 0, stream>>>(
        mb_xp_W + (long long)i * D_INNER_ * 2 * D_STATE_, ws + XPW, D_INNER_, 2 * D_STATE_, 64);
    pad_cols_kernel<<<1, 64, 0, stream>>>(mb_xp_b + i * 2 * D_STATE_, ws + XPB, 1, 2 * D_STATE_, 64);
    // [Bs|Cs|0pad]: 2048 x 1024 -> 64 (padded)
    gemm(ws + XCV, D_INNER_, 0, 0, ws + XPW, 64, 0, 0, 0, ws + XPB,
         ws + BCS, 64, 0, 0, MROWS, 64, D_INNER_, 1, 1, 1.0f, 0, 0);
    // delta = softplus(xc Wdt + bdt): 2048 x 1024 -> 1024
    gemm(ws + XCV, D_INNER_, 0, 0, mb_dt_W + (long long)i * D_INNER_ * D_INNER_,
         D_INNER_, 0, 0, 0, mb_dt_b + i * D_INNER_,
         ws + DLT, D_INNER_, 0, 0, MROWS, D_INNER_, D_INNER_, 1, 1, 1.0f, 3, 0);
    // selective scan + D skip + z gate
    mamba_scan_kernel<<<(B_ * D_INNER_) / 256, 256, 0, stream>>>(
        ws + XCV, ws + DLT, ws + BCS, ws + MIN,
        mb_Alog + (long long)i * D_INNER_ * D_STATE_, mb_D + i * D_INNER_, ws + YB);
    // x += y Wout + bout
    gemm(ws + YB, D_INNER_, 0, 0, mb_out_W + (long long)i * D_INNER_ * D_MODEL_,
         D_MODEL_, 0, 0, 0, mb_out_b + i * D_MODEL_,
         ws + XM, D_MODEL_, 0, 0, MROWS, D_MODEL_, D_INNER_, 1, 1, 1.0f, 0, 1);
  }

  // --------------------------------------------------------------------- head
  // pad head weights/bias 41 -> 64 cols, GEMM into padded buffer, then unpad
  pad_cols_kernel<<<(D_MODEL_ * 64) / 256, 256, 0, stream>>>(head_W, ws + HWP, D_MODEL_, NPH_, 64);
  pad_cols_kernel<<<1, 64, 0, stream>>>(head_b, ws + HBP, 1, NPH_, 64);
  gemm(ws + XM, D_MODEL_, 0, 0, ws + HWP, 64, 0, 0, 0, ws + HBP,
       ws + OUTP, 64, 0, 0, MROWS, 64, D_MODEL_, 1, 1, 1.0f, 0, 0);
  unpad_cols_kernel<<<(MROWS * NPH_ + 255) / 256, 256, 0, stream>>>(
      ws + OUTP, (float*)d_out, MROWS, 64, NPH_);
}
